// ModeConditionedSparseMoE_75007308857547
// MI455X (gfx1250) — matrix operand order repaired and checked by reading, not compiled, not run on hardware
//
#include <hip/hip_runtime.h>
#include <cstddef>

// ---------------------------------------------------------------------------
// Mode-conditioned sparse MoE for MI455X (gfx1250), bf16 WMMA path.
// D=1024, H=2048, E=8 routed, M=4 modes, S=1 shared, top-2, T=4096 tokens.
// ---------------------------------------------------------------------------

typedef __bf16 bf16_t;
typedef __attribute__((ext_vector_type(16))) __bf16 v16bf;
typedef __attribute__((ext_vector_type(8)))  __bf16 v8bf;
typedef __attribute__((ext_vector_type(8)))  float  v8f;

#define DMODEL 1024
#define HDIM   2048
#define NEXP   8
#define NMODE  4
#define NTOK   4096
#define NSLOT  13              // 1 shared + 4 mode + 8 routed
#define TILE_M 32
#define KT1 (DMODEL/32)        // 32 k-tiles for GEMM1
#define KT2 (HDIM/32)          // 64 k-tiles for GEMM2
#define TILES_PER_SLOT (NTOK/TILE_M)   // 128

// fp32 -> bf16 round-to-nearest-even (explicit, no runtime-lib dependence)
__device__ __forceinline__ bf16_t f2bf(float f) {
  unsigned u = __float_as_uint(f);
  unsigned short h = (unsigned short)((u + 0x7FFFu + ((u >> 16) & 1u)) >> 16);
  return __builtin_bit_cast(bf16_t, h);
}

// jax.nn.gelu default = tanh approximation
__device__ __forceinline__ float gelu_tanh(float x) {
  float x3 = x * x * x;
  return 0.5f * x * (1.f + tanhf(0.7978845608028654f * (x + 0.044715f * x3)));
}

// ---------------------------------------------------------------------------
// Kernel 1: weight pack fp32 -> bf16 in WMMA B-operand tile layout.
// B tile = 32(K) x 16(N): lane<16 -> N=lane, K=0..7,16..23; lane>=16 -> N=lane-16,
// K=8..15,24..31; per lane 16 contiguous bf16 (32 bytes) => one vector load later.
// Tiles stored [slot][ntile][ktile][lane][16].
// ---------------------------------------------------------------------------
__global__ void pack_weights(const float* __restrict__ sw1,
                             const float* __restrict__ mw1,
                             const float* __restrict__ rw1,
                             const float* __restrict__ sw2,
                             const float* __restrict__ mw2,
                             const float* __restrict__ rw2,
                             bf16_t* __restrict__ w1t,
                             bf16_t* __restrict__ w2t) {
  size_t id = (size_t)blockIdx.x * blockDim.x + threadIdx.x;
  const size_t per = (size_t)DMODEL * HDIM;          // halves per expert slot
  const size_t tot = (size_t)NSLOT * per;
  if (id >= 2 * tot) return;
  bool isW2 = id >= tot;
  size_t r = isW2 ? id - tot : id;
  int slot = (int)(r / per);
  size_t q = r % per;
  int tile = (int)(q >> 9);                          // /512 halves per tile
  int lane = (int)((q >> 4) & 31);
  int j    = (int)(q & 15);
  int KT   = isW2 ? KT2 : KT1;
  int kt   = tile % KT;
  int nt   = tile / KT;
  int n = nt * 16 + (lane & 15);
  int k = kt * 32 + ((j >> 3) << 4) + ((lane >> 4) << 3) + (j & 7);  // K: 8h+j, 16+8h+(j-8)
  float v;
  if (!isW2) {                                        // w1: [Dmodel, H], k in D, n in H
    const float* src = (slot == 0) ? sw1
                     : (slot < 5)  ? mw1 + (size_t)(slot - 1) * per
                                   : rw1 + (size_t)(slot - 5) * per;
    v = src[(size_t)k * HDIM + n];
    w1t[r] = f2bf(v);
  } else {                                            // w2: [H, Dmodel], k in H, n in D
    const float* src = (slot == 0) ? sw2
                     : (slot < 5)  ? mw2 + (size_t)(slot - 1) * per
                                   : rw2 + (size_t)(slot - 5) * per;
    v = src[(size_t)k * DMODEL + n];
    w2t[r] = f2bf(v);
  }
}

// ---------------------------------------------------------------------------
// Kernel 2: router (logits, softmax, top-2), writes combine weights + aux.
// ---------------------------------------------------------------------------
__global__ void router_kernel(const float* __restrict__ hidden,
                              const float* __restrict__ rw,
                              const float* __restrict__ rb,
                              float* __restrict__ combine,
                              int* __restrict__ top0,
                              float* __restrict__ d_out) {
  int t = blockIdx.x * blockDim.x + threadIdx.x;
  if (t >= NTOK) return;
  float l[NEXP];
  #pragma unroll
  for (int e = 0; e < NEXP; ++e) l[e] = rb[e];
  const float* x = hidden + (size_t)t * DMODEL;
  for (int d = 0; d < DMODEL; ++d) {
    float xv = x[d];
    const float* w = rw + (size_t)d * NEXP;
    #pragma unroll
    for (int e = 0; e < NEXP; ++e) l[e] += xv * w[e];
  }
  float mx = l[0];
  #pragma unroll
  for (int e = 1; e < NEXP; ++e) mx = fmaxf(mx, l[e]);
  float p[NEXP], s = 0.f;
  #pragma unroll
  for (int e = 0; e < NEXP; ++e) { p[e] = __expf(l[e] - mx); s += p[e]; }
  float inv = 1.f / s;
  #pragma unroll
  for (int e = 0; e < NEXP; ++e) p[e] *= inv;
  int i0 = 0;
  #pragma unroll
  for (int e = 1; e < NEXP; ++e) if (p[e] > p[i0]) i0 = e;   // ties -> lowest idx
  int i1 = (i0 == 0) ? 1 : 0;
  #pragma unroll
  for (int e = 0; e < NEXP; ++e) if (e != i0 && p[e] > p[i1]) i1 = e;
  #pragma unroll
  for (int e = 0; e < NEXP; ++e)
    combine[(size_t)t * NEXP + e] = (e == i0) ? p[i0] : (e == i1) ? p[i1] : 0.f;
  top0[t] = i0;
  // aux outputs, concatenated after the main output in d_out
  float* alog = d_out + (size_t)NTOK * DMODEL;
  float* aidx = alog + (size_t)NTOK * NEXP;
  float* aprb = aidx + (size_t)NTOK * 2;
  #pragma unroll
  for (int e = 0; e < NEXP; ++e) alog[(size_t)t * NEXP + e] = l[e];
  aidx[t * 2 + 0] = (float)i0;
  aidx[t * 2 + 1] = (float)i1;
  aprb[t * 2 + 0] = p[i0];
  aprb[t * 2 + 1] = p[i1];
}

// ---------------------------------------------------------------------------
// Kernel 3: deterministic per-slot token-list compaction (block-wide scan,
// no atomics => bitwise-stable list order => deterministic output).
// Slot 0 = shared, 1..4 = modes, 5..12 = routed experts.
// ---------------------------------------------------------------------------
__global__ void build_lists(const int* __restrict__ mode_ids,
                            const float* __restrict__ combine,
                            int* __restrict__ lists,
                            int* __restrict__ counts) {
  __shared__ int scan[256];
  __shared__ int base;
  int slot = blockIdx.x;
  int tid  = threadIdx.x;
  if (tid == 0) base = 0;
  __syncthreads();
  for (int chunk = 0; chunk < NTOK; chunk += 256) {
    int t = chunk + tid;
    int flag;
    if (slot == 0)      flag = 1;
    else if (slot < 5)  flag = (mode_ids[t] == slot - 1);
    else                flag = (combine[(size_t)t * NEXP + (slot - 5)] > 0.f);
    scan[tid] = flag;
    __syncthreads();
    for (int off = 1; off < 256; off <<= 1) {       // inclusive Hillis-Steele
      int v = (tid >= off) ? scan[tid - off] : 0;
      __syncthreads();
      scan[tid] += v;
      __syncthreads();
    }
    if (flag) lists[(size_t)slot * NTOK + base + scan[tid] - 1] = t;
    __syncthreads();
    if (tid == 0) base += scan[255];
    __syncthreads();
  }
  if (tid == 0) counts[slot] = base;
}

// ---------------------------------------------------------------------------
// Kernel 4: fused expert FFN, bf16 WMMA.
// One workgroup = (slot, 32-token tile). 8 waves of 32 lanes.
// LDS: X tile 32x1024 bf16 (64KB) + h tile 32x2048 bf16 (128KB) + row meta.
// GEMM1: h = gelu(X @ W1 + b1); GEMM2: y = scale * (h @ W2 + b2) -> plane.
// ---------------------------------------------------------------------------
__global__ __launch_bounds__(256)
void moe_ffn(const float* __restrict__ hidden,
             const bf16_t* __restrict__ w1t,
             const bf16_t* __restrict__ w2t,
             const float* __restrict__ shared_b1,
             const float* __restrict__ mode_b1,
             const float* __restrict__ routed_b1,
             const float* __restrict__ shared_b2,
             const float* __restrict__ mode_b2,
             const float* __restrict__ routed_b2,
             const float* __restrict__ combine,
             const int* __restrict__ top0,
             const int* __restrict__ lists,
             const int* __restrict__ counts,
             float* __restrict__ planes) {
  extern __shared__ char smem[];
  bf16_t* Xt = (bf16_t*)smem;                              // 65536 B
  bf16_t* Ht = (bf16_t*)(smem + 65536);                    // 131072 B
  int*    tok    = (int*)  (smem + 65536 + 131072);        // 128 B
  float*  rscale = (float*)(smem + 65536 + 131072 + 128);  // 128 B
  int*    rplane = (int*)  (smem + 65536 + 131072 + 256);  // 128 B

  int slot = blockIdx.x / TILES_PER_SLOT;
  int tile = blockIdx.x % TILES_PER_SLOT;
  int cnt  = counts[slot];
  if (tile * TILE_M >= cnt) return;                        // uniform exit

  const float* b1;
  const float* b2;
  if (slot == 0)      { b1 = shared_b1;                          b2 = shared_b2; }
  else if (slot < 5)  { b1 = mode_b1  + (size_t)(slot - 1) * HDIM;
                        b2 = mode_b2  + (size_t)(slot - 1) * DMODEL; }
  else                { b1 = routed_b1 + (size_t)(slot - 5) * HDIM;
                        b2 = routed_b2 + (size_t)(slot - 5) * DMODEL; }

  int tid = threadIdx.x;
  if (tid < TILE_M) {
    int g = tile * TILE_M + tid;
    if (g < cnt) {
      int t = lists[(size_t)slot * NTOK + g];
      tok[tid] = t;
      if (slot == 0)      { rscale[tid] = 1.f; rplane[tid] = 0; }
      else if (slot < 5)  { rscale[tid] = 1.f; rplane[tid] = 1; }
      else {
        int e = slot - 5;
        rscale[tid] = combine[(size_t)t * NEXP + e];
        rplane[tid] = (top0[t] == e) ? 2 : 3;
      }
    } else {
      tok[tid] = 0; rscale[tid] = 0.f; rplane[tid] = -1;   // pad row: no store
    }
  }
  __syncthreads();

  // Gather X tile (coalesced: consecutive threads -> consecutive columns)
  for (int i = tid; i < TILE_M * DMODEL; i += 256) {
    int r = i >> 10, c = i & (DMODEL - 1);
    Xt[i] = f2bf(hidden[(size_t)tok[r] * DMODEL + c]);
  }
  __syncthreads();

  const int lane    = tid & 31;
  const int wv      = tid >> 5;
  const int halfrow = lane >> 4;      // 0 | 1 (selects A K-half / C row-half)
  const int lm      = lane & 15;

  // ---- GEMM1: 32x2048 = X(32x1024) @ W1(1024x2048), +b1, gelu -> Ht ----
  const v16bf* w1v = (const v16bf*)(w1t + (size_t)slot * DMODEL * HDIM);
  for (int mt = 0; mt < 2; ++mt) {
    const bf16_t* aBase = Xt + (size_t)(mt * 16 + lm) * DMODEL + halfrow * 8;
    for (int nl = 0; nl < 16; ++nl) {
      int nt = wv * 16 + nl;                         // 8 waves x 16 = 128 ntiles
      v8f acc = {};
      for (int kt = 0; kt < KT1; ++kt) {
        // A fragment (16x32 bf16): lane holds row lm; halves K b..b+7, b+16..b+23
        v8bf a0 = *(const v8bf*)(aBase + kt * 32);
        v8bf a1 = *(const v8bf*)(aBase + kt * 32 + 16);
        v16bf a;
        #pragma unroll
        for (int q = 0; q < 8; ++q) { a[q] = a0[q]; a[q + 8] = a1[q]; }
        // B fragment: one 32-byte per-lane load from packed tiles
        v16bf b = w1v[((size_t)nt * KT1 + kt) * 32 + lane];
        acc = __builtin_amdgcn_wmma_f32_16x16x32_bf16(
            false, a, false, b, (short)0, acc, false, false);
      }
      int n = nt * 16 + lm;
      float bv = b1[n];
      #pragma unroll
      for (int q = 0; q < 8; ++q) {                  // C/D: row = q + halfrow*8
        int row = mt * 16 + halfrow * 8 + q;
        Ht[(size_t)row * HDIM + n] = f2bf(gelu_tanh(acc[q] + bv));
      }
    }
  }
  __syncthreads();

  // ---- GEMM2: 32x1024 = Ht(32x2048) @ W2(2048x1024), +b2, scale -> plane ----
  const v16bf* w2v = (const v16bf*)(w2t + (size_t)slot * DMODEL * HDIM);
  for (int mt = 0; mt < 2; ++mt) {
    const bf16_t* aBase = Ht + (size_t)(mt * 16 + lm) * HDIM + halfrow * 8;
    for (int nl = 0; nl < 8; ++nl) {
      int nt = wv * 8 + nl;                          // 8 waves x 8 = 64 ntiles
      v8f acc = {};
      for (int kt = 0; kt < KT2; ++kt) {
        v8bf a0 = *(const v8bf*)(aBase + kt * 32);
        v8bf a1 = *(const v8bf*)(aBase + kt * 32 + 16);
        v16bf a;
        #pragma unroll
        for (int q = 0; q < 8; ++q) { a[q] = a0[q]; a[q + 8] = a1[q]; }
        v16bf b = w2v[((size_t)nt * KT2 + kt) * 32 + lane];
        acc = __builtin_amdgcn_wmma_f32_16x16x32_bf16(
            false, a, false, b, (short)0, acc, false, false);
      }
      int n = nt * 16 + lm;
      float bv = b2[n];
      #pragma unroll
      for (int q = 0; q < 8; ++q) {
        int row = mt * 16 + halfrow * 8 + q;
        int pl  = rplane[row];
        if (pl >= 0) {
          planes[(size_t)pl * NTOK * DMODEL + (size_t)tok[row] * DMODEL + n] =
              rscale[row] * (acc[q] + bv);
        }
      }
    }
  }
}

// ---------------------------------------------------------------------------
// Kernel 5: out = shared + mode + routed_top1 + routed_top2 (every element of
// every plane is written exactly once above, so no pre-zeroing needed).
// ---------------------------------------------------------------------------
__global__ void reduce_planes(const float* __restrict__ planes,
                              float* __restrict__ out) {
  size_t i = (size_t)blockIdx.x * blockDim.x + threadIdx.x;
  const size_t N = (size_t)NTOK * DMODEL;
  if (i < N)
    out[i] = planes[i] + planes[N + i] + planes[2 * N + i] + planes[3 * N + i];
}

// ---------------------------------------------------------------------------
extern "C" void kernel_launch(void* const* d_in, const int* in_sizes, int n_in,
                              void* d_out, int out_size, void* d_ws, size_t ws_size,
                              hipStream_t stream) {
  (void)in_sizes; (void)n_in; (void)out_size; (void)ws_size;
  const float* hidden    = (const float*)d_in[0];
  const int*   mode_ids  = (const int*)  d_in[1];
  const float* router_w  = (const float*)d_in[2];
  const float* router_b  = (const float*)d_in[3];
  const float* shared_w1 = (const float*)d_in[4];
  const float* shared_b1 = (const float*)d_in[5];
  const float* shared_w2 = (const float*)d_in[6];
  const float* shared_b2 = (const float*)d_in[7];
  const float* mode_w1   = (const float*)d_in[8];
  const float* mode_b1   = (const float*)d_in[9];
  const float* mode_w2   = (const float*)d_in[10];
  const float* mode_b2   = (const float*)d_in[11];
  const float* routed_w1 = (const float*)d_in[12];
  const float* routed_b1 = (const float*)d_in[13];
  const float* routed_w2 = (const float*)d_in[14];
  const float* routed_b2 = (const float*)d_in[15];
  float* out = (float*)d_out;

  // workspace carve-up (~174 MB total)
  char* ws = (char*)d_ws;
  const size_t perSlotHalves = (size_t)DMODEL * HDIM;
  const size_t w1t_bytes = (size_t)NSLOT * perSlotHalves * sizeof(bf16_t); // 54.5MB
  bf16_t* w1t   = (bf16_t*)(ws);
  bf16_t* w2t   = (bf16_t*)(ws + w1t_bytes);
  size_t off    = 2 * w1t_bytes;
  float* combine = (float*)(ws + off); off += (size_t)NTOK * NEXP * 4;      // 128KB
  int*   top0    = (int*)  (ws + off); off += (size_t)NTOK * 4;             // 16KB
  int*   lists   = (int*)  (ws + off); off += (size_t)NSLOT * NTOK * 4;     // 208KB
  int*   counts  = (int*)  (ws + off); off += 256;
  float* planes  = (float*)(ws + off);                                      // 64MB

  // 1) pack weights to bf16 WMMA tiles
  {
    size_t total = 2 * (size_t)NSLOT * perSlotHalves;   // 54,525,952
    dim3 g((unsigned)((total + 255) / 256));
    pack_weights<<<g, 256, 0, stream>>>(shared_w1, mode_w1, routed_w1,
                                        shared_w2, mode_w2, routed_w2,
                                        w1t, w2t);
  }
  // 2) router + aux outputs
  router_kernel<<<NTOK / 256, 256, 0, stream>>>(hidden, router_w, router_b,
                                                combine, top0, out);
  // 3) deterministic token lists
  build_lists<<<NSLOT, 256, 0, stream>>>(mode_ids, combine, lists, counts);
  // 4) fused expert FFNs (bf16 WMMA), 197KB dynamic LDS per workgroup
  {
    const size_t smem = 65536 + 131072 + 3 * 128;       // 196,992 B < 320KB
    moe_ffn<<<NSLOT * TILES_PER_SLOT, 256, smem, stream>>>(
        hidden, w1t, w2t,
        shared_b1, mode_b1, routed_b1,
        shared_b2, mode_b2, routed_b2,
        combine, top0, lists, counts, planes);
  }
  // 5) final reduce into d_out
  reduce_planes<<<(NTOK * DMODEL) / 256, 256, 0, stream>>>(planes, out);
}
